// FNO1d_70815420776999
// MI455X (gfx1250) — matrix operand order repaired and checked by reading, not compiled, not run on hardware
//
#include <hip/hip_runtime.h>

#define Bb    16
#define Nn    16384
#define W     64
#define MODES 32
#define NBLK  4

typedef __attribute__((ext_vector_type(16))) _Float16 v16h;
typedef __attribute__((ext_vector_type(8)))  _Float16 v8h;
typedef __attribute__((ext_vector_type(8)))  float    v8f;
typedef unsigned int u32;
typedef __attribute__((ext_vector_type(4))) u32 u32x4;
typedef __attribute__((ext_vector_type(8))) u32 u32x8;
typedef _Float16 f16;

// D = A(16x32 f16) * B(32x16 f16) + C(16x16 f32)   [CDNA5 wave32 WMMA]
__device__ __forceinline__ v8f wmma16(v16h a, v16h b, v8f c) {
  return __builtin_amdgcn_wmma_f32_16x16x32_f16(false, a, false, b, (short)0, c,
                                                false, false);
}

__device__ __forceinline__ v16h cat8(v8h lo, v8h hi) {
  return __builtin_shufflevector(lo, hi, 0, 1, 2, 3, 4, 5, 6, 7,
                                         8, 9, 10, 11, 12, 13, 14, 15);
}

// Scale split so both coeffs and basis stay in f16 normal range:
// invSqrtN * invSqrtN = 1/N ; r2n * r2n = 2/N
#define INV_SQRT_N 0.0078125f            /* 1/128 = 1/sqrt(16384) */
#define R2N        0.011048543456039804f /* sqrt(2/16384)          */

// ---------------------------------------------------------------------------
// Truncated-DFT bases, written DIRECTLY in WMMA B-fragment order so that each
// lane's 16 f16 elements are one contiguous, 32B-aligned v16h load.
//
// Forward bFs: fragment (ktile=k/32, jt=j/16); lane = 16*(kin>=16) + (j&15);
//              element e = kin&15;  value: col 2m = cos, col 2m+1 = -sin.
// Inverse bIs: fragment (ntile=n/16, s=j/32); lane = 16*((j&31)>=16)+(n&15);
//              element e = j&15;    rows: 0 = invSqrtN, 2m-1 = r2n*cos,
//              2m = -r2n*sin, 63 = 0.
// ---------------------------------------------------------------------------
__global__ void k_gen_basis(f16* __restrict__ bFs, f16* __restrict__ bIs) {
  int n = blockIdx.x * blockDim.x + threadIdx.x;
  if (n >= Nn) return;
  const float w0 = 6.2831853071795864769f / (float)Nn;
  int ktile = n >> 5, kin = n & 31;
  int lbase = (kin >> 4) << 4;           // forward lane-half base
  int eF = kin & 15;
  int ntile = n >> 4, lnI = n & 15;

  for (int m = 0; m < MODES; ++m) {
    int idx = (m * n) & (Nn - 1);        // exact arg reduction: (m*n) mod N
    float s, c;
    __sincosf(w0 * (float)idx, &s, &c);

    // forward columns j0 = 2m (cos), j1 = 2m+1 (-sin)
    int j0 = 2 * m, j1 = 2 * m + 1;
    bFs[((size_t)(ktile * 4 + (j0 >> 4)) * 32 + lbase + (j0 & 15)) * 16 + eF] = (f16)c;
    bFs[((size_t)(ktile * 4 + (j1 >> 4)) * 32 + lbase + (j1 & 15)) * 16 + eF] = (f16)(-s);

    // inverse rows
    if (m == 0) {
      // j = 0 : s=0, lanehalf=0, e=0
      bIs[((size_t)(ntile * 2 + 0) * 32 + 0 + lnI) * 16 + 0]  = (f16)INV_SQRT_N;
      // j = 63 : s=1, kin=31 -> lanehalf=1, e=15
      bIs[((size_t)(ntile * 2 + 1) * 32 + 16 + lnI) * 16 + 15] = (f16)0.0f;
    } else {
      int ja = 2 * m - 1, jb = 2 * m;
      int sa = ja >> 5, ka = ja & 31;
      int sb = jb >> 5, kb2 = jb & 31;
      bIs[((size_t)(ntile * 2 + sa) * 32 + ((ka >> 4) << 4) + lnI) * 16 + (ka & 15)]
          = (f16)(R2N * c);
      bIs[((size_t)(ntile * 2 + sb) * 32 + ((kb2 >> 4) << 4) + lnI) * 16 + (kb2 & 15)]
          = (f16)(-R2N * s);
    }
  }
}

// h16[b][o][n] = f16( x[b][0][n]*lift_w[o] + lift_b[o] )
__global__ void k_lift(const float* __restrict__ x, const float* __restrict__ lw,
                       const float* __restrict__ lb, f16* __restrict__ h16) {
  size_t i = (size_t)blockIdx.x * blockDim.x + threadIdx.x;
  int n = (int)(i & (Nn - 1));
  int r = (int)(i >> 14);
  int o = r & (W - 1);
  int b = r >> 6;
  h16[i] = (f16)(x[(size_t)b * Nn + n] * lw[o] + lb[o]);
}

// ---------------------------------------------------------------------------
// Forward truncated DFT: X[b][i][0:64] = sum_n h16[b][i][n] * basisF[n][:]
// Grid: Bb*4 WGs (b, 16-row tile) x 256 threads (8 waves).
// Split-K: wave w covers n in [w*2048,(w+1)*2048); LDS cross-wave reduce.
// A fragments: two contiguous b128 loads (h16 row-major).
// B fragments: one 32B v16h load from pre-swizzled bFs.
// ---------------------------------------------------------------------------
__global__ void __launch_bounds__(256) k_fwd_dft(const f16* __restrict__ h16,
                                                 const f16* __restrict__ bFs,
                                                 float* __restrict__ X) {
  __shared__ float red[16 * 64];
  int b    = blockIdx.x >> 2;
  int i0   = (blockIdx.x & 3) * 16;
  int wave = threadIdx.x >> 5;
  int lane = threadIdx.x & 31;
  int lh   = lane >> 4;
  int ll   = lane & 15;
  int kb   = lh * 16;

  const f16* hr = h16 + (size_t)(b * W + i0 + ll) * Nn;   // A row M = ll
  v8f acc[4] = {};

  int kend = wave * 2048 + 2048;
  for (int kk = wave * 2048; kk < kend; kk += 32) {
    __builtin_prefetch(hr + kk + 256, 0, 0);              // global_prefetch_b8
    v8h alo = *(const v8h*)(hr + kk + kb);
    v8h ahi = *(const v8h*)(hr + kk + 8 + kb);
    v16h a = cat8(alo, ahi);
    const f16* bp = bFs + (size_t)(kk >> 5) * 4 * 512 + lane * 16;
#pragma unroll
    for (int jt = 0; jt < 4; ++jt) {
      v16h bb = *(const v16h*)(bp + jt * 512);
      acc[jt] = wmma16(a, bb, acc[jt]);
    }
  }

  // Cross-wave reduction in LDS (serialized add phases).
  for (int p = 0; p < 8; ++p) {
    if (wave == p) {
#pragma unroll
      for (int jt = 0; jt < 4; ++jt)
#pragma unroll
        for (int v = 0; v < 8; ++v) {
          int m = v + lh * 8;            // C-layout: M = v + 8*(lane>=16)
          int j = jt * 16 + ll;
          if (p == 0) red[m * 64 + j]  = acc[jt][v];
          else        red[m * 64 + j] += acc[jt][v];
        }
    }
    __syncthreads();
  }
  for (int idx = threadIdx.x; idx < 1024; idx += 256) {
    int m = idx >> 6, j = idx & 63;
    X[(size_t)(b * W + i0 + m) * 64 + j] = red[idx];
  }
}

// ---------------------------------------------------------------------------
// Complex mode mixing + scaling into f16 inverse-basis coefficients.
// C[b][o][0]=Yr0*invSqrtN ; C[2m-1]=Yr*r2n ; C[2m]=Yi*r2n ; C[63]=0
// ---------------------------------------------------------------------------
__global__ void k_modemix(const float* __restrict__ X, const float* __restrict__ wr,
                          const float* __restrict__ wi, f16* __restrict__ C, int k) {
  int t = blockIdx.x * blockDim.x + threadIdx.x;   // Bb*W*MODES = 32768
  int m = t & 31;
  int o = (t >> 5) & 63;
  int b = t >> 11;
  const float* Xb  = X  + (size_t)b * W * 64;
  const float* wrk = wr + (size_t)k * W * W * MODES;
  const float* wik = wi + (size_t)k * W * W * MODES;
  float yr = 0.f, yi = 0.f;
  for (int i = 0; i < W; ++i) {
    float xr = Xb[i * 64 + 2 * m];
    float xi = Xb[i * 64 + 2 * m + 1];
    float a  = wrk[((size_t)i * W + o) * MODES + m];   // wr[k][i][o][m]
    float c2 = wik[((size_t)i * W + o) * MODES + m];
    yr += xr * a - xi * c2;
    yi += xr * c2 + xi * a;
  }
  f16* Cb = C + (size_t)(b * W + o) * 64;
  if (m == 0) { Cb[0] = (f16)(yr * INV_SQRT_N); Cb[63] = (f16)0.f; }
  else        { Cb[2 * m - 1] = (f16)(yr * R2N); Cb[2 * m] = (f16)(yi * R2N); }
}

// ---------------------------------------------------------------------------
// Fused inverse-DFT + bypass + bias, IN PLACE on h16.
// h16[b][o][n] = sum_j C[b][o][j]*bI[j][n] + sum_i byp_w[k][o][i]*h16[b][i][n]
//               + byp_b[k][o]
// Grid: Bb*(Nn/128) WGs x 256 threads; wave owns a 16-column subtile.
// Spectral coeff matrix C[b] (8 KB f16) is DMA'd into LDS by the Tensor Data
// Mover (tensor_load_to_lds, D# per CDNA5 ISA), byp_w is
// loaded+converted cooperatively. Each wave reads all of its columns before
// writing them -> in-place safe.
// ---------------------------------------------------------------------------
__global__ void __launch_bounds__(256) k_inv_bypass(f16* __restrict__ h16,
    const f16* __restrict__ C, const f16* __restrict__ bIs,
    const float* __restrict__ bw, const float* __restrict__ bbias, int k) {
  __shared__ f16 Asc[64 * 64];   // spectral coeffs C[b]  (rows o, cols j)
  __shared__ f16 Abp[64 * 64];   // byp_w[k]              (rows o, cols i)
  int b    = blockIdx.x >> 7;
  int tile = blockIdx.x & 127;

  // --- TDM: async DMA C[b] (4096 x f16, 1-D tile) into Asc --------------
  if (threadIdx.x < 32) {        // wave 0 issues; TDM ignores EXEC
    unsigned long long ga = (unsigned long long)(const void*)(C + (size_t)b * 4096);
    u32 lds = (u32)(unsigned long long)(void*)&Asc[0];
    u32x4 g0;
    g0.x = 1u;                                   // count=1, user mode
    g0.y = lds;                                  // lds_addr
    g0.z = (u32)ga;                              // global_addr[31:0]
    g0.w = (u32)(ga >> 32) | (2u << 30);         // global_addr[56:32] | type=2
    u32x8 g1;
    g1.s0 = 1u << 16;                            // data_size=1 -> 2 bytes
    g1.s1 = 4096u << 16;                         // tensor_dim0 = 4096 (lo16)
    g1.s2 = 1u << 16;                            // tensor_dim1 = 1
    g1.s3 = 4096u << 16;                         // tile_dim0 = 4096
    g1.s4 = 1u;                                  // tile_dim1 = 1
    g1.s5 = 4096u;                               // tensor_dim0_stride = 4096
    g1.s6 = 0u;
    g1.s7 = 0u;
    asm volatile("tensor_load_to_lds %0, %1" :: "s"(g0), "s"(g1) : "memory");
    __builtin_amdgcn_s_wait_tensorcnt(0);
  }
  const float* bwk = bw + (size_t)k * 4096;
  for (int idx = threadIdx.x; idx < 4096; idx += 256)
    Abp[idx] = (f16)bwk[idx];
  __syncthreads();

  int wave = threadIdx.x >> 5, lane = threadIdx.x & 31;
  int lh = lane >> 4, ll = lane & 15;
  int kb = lh * 16, khalf = lh * 16;
  int ncol = tile * 128 + wave * 16 + ll;
  int ntile = tile * 8 + wave;

  // B fragments for the inverse DFT: pre-swizzled, one 32B load per K-half.
  const f16* bIb = bIs + (size_t)ntile * 2 * 512 + lane * 16;
  v16h bsc[2];
  bsc[0] = *(const v16h*)(bIb);
  bsc[1] = *(const v16h*)(bIb + 512);

  v8f acc[4] = {};
#pragma unroll
  for (int s = 0; s < 2; ++s) {
    int k0 = s * 32;
    // bypass B fragment: gathered u16 from h16 (stride Nn), no conversion
    v16h bbp;
#pragma unroll
    for (int e = 0; e < 16; ++e) {
      int K = k0 + khalf + e;
      bbp[e] = h16[(size_t)(b * W + K) * Nn + ncol];
    }
#pragma unroll
    for (int t = 0; t < 4; ++t) {
      int row = t * 16 + ll;
      v16h a1 = cat8(*(const v8h*)&Asc[row * 64 + k0 + kb],
                     *(const v8h*)&Asc[row * 64 + k0 + 8 + kb]);
      v16h a2 = cat8(*(const v8h*)&Abp[row * 64 + k0 + kb],
                     *(const v8h*)&Abp[row * 64 + k0 + 8 + kb]);
      acc[t] = wmma16(a1, bsc[s], acc[t]);
      acc[t] = wmma16(a2, bbp, acc[t]);
    }
  }

  const float* bbk = bbias + (size_t)k * W;
#pragma unroll
  for (int t = 0; t < 4; ++t)
#pragma unroll
    for (int v = 0; v < 8; ++v) {
      int o = t * 16 + lh * 8 + v;
      h16[(size_t)(b * W + o) * Nn + ncol] = (f16)(acc[t][v] + bbk[o]);
    }
}

// out[b][0][n] = sum_w h16[b][w][n]*proj_w[w] + proj_b[0]
__global__ void k_proj(const f16* __restrict__ h16, const float* __restrict__ pw,
                       const float* __restrict__ pb, float* __restrict__ out) {
  int t = blockIdx.x * blockDim.x + threadIdx.x;   // Bb*Nn = 262144
  int n = t & (Nn - 1);
  int b = t >> 14;
  float acc = pb[0];
  for (int w = 0; w < W; ++w)
    acc += (float)h16[(size_t)(b * W + w) * Nn + n] * pw[w];
  out[(size_t)b * Nn + n] = acc;
}

extern "C" void kernel_launch(void* const* d_in, const int* in_sizes, int n_in,
                              void* d_out, int out_size, void* d_ws, size_t ws_size,
                              hipStream_t stream) {
  (void)in_sizes; (void)n_in; (void)out_size; (void)ws_size;
  const float* x  = (const float*)d_in[0];
  const float* lw = (const float*)d_in[1];
  const float* lb = (const float*)d_in[2];
  const float* wr = (const float*)d_in[3];
  const float* wi = (const float*)d_in[4];
  const float* bw = (const float*)d_in[5];
  const float* bb = (const float*)d_in[6];
  const float* pw = (const float*)d_in[7];
  const float* pb = (const float*)d_in[8];
  float* out = (float*)d_out;

  char* ws = (char*)d_ws;
  f16*   h16 = (f16*)  (ws);                // 16*64*16384 f16 = 32 MB
  float* X   = (float*)(ws + 33554432);     // 16*64*64 f32    = 256 KB
  f16*   C   = (f16*)  (ws + 33816576);     // 16*64*64 f16    = 128 KB
  f16*   bFs = (f16*)  (ws + 33947648);     // 16384*64 f16    = 2 MB (swizzled)
  f16*   bIs = (f16*)  (ws + 36044800);     // 64*16384 f16    = 2 MB (swizzled)
                                            // total ~36.4 MB (L2-resident)

  k_gen_basis<<<Nn / 128, 128, 0, stream>>>(bFs, bIs);
  k_lift<<<(Bb * W * Nn) / 256, 256, 0, stream>>>(x, lw, lb, h16);
  for (int k = 0; k < NBLK; ++k) {
    k_fwd_dft<<<Bb * 4, 256, 0, stream>>>(h16, bFs, X);
    k_modemix<<<(Bb * W * MODES) / 256, 256, 0, stream>>>(X, wr, wi, C, k);
    k_inv_bypass<<<Bb * (Nn / 128), 256, 0, stream>>>(h16, C, bIs, bw, bb, k);
  }
  k_proj<<<(Bb * Nn) / 256, 256, 0, stream>>>(h16, pw, pb, out);
}
